// TinySelfAttention_19593640804686
// MI455X (gfx1250) — matrix operand order repaired and checked by reading, not compiled, not run on hardware
//
#include <hip/hip_runtime.h>
#include <hip/hip_bf16.h>

// ---------------------------------------------------------------------------
// TinySelfAttention on MI455X (gfx1250): bf16 WMMA everywhere, fp32 accum.
// T = D = 2048, H = 16, DH = 128.
// GEMMs: block tile 128x64, B panel staged to LDS via Tensor Data Mover
// (double buffered, s_wait_tensorcnt), A reused across 4 WMMAs per wave.
// ---------------------------------------------------------------------------

typedef __bf16 bf16;
typedef __attribute__((ext_vector_type(16))) __bf16 v16bf;
typedef __attribute__((ext_vector_type(8)))  __bf16 v8bf;
typedef __attribute__((ext_vector_type(4)))  __bf16 v4bf;
typedef __attribute__((ext_vector_type(8)))  float  v8f;

typedef __attribute__((ext_vector_type(4))) unsigned int u32x4;
typedef __attribute__((ext_vector_type(8))) int          i32x8;
typedef __attribute__((ext_vector_type(4))) int          i32x4;

#define TT 2048
#define DD 2048
#define NH 16
#define DH 128

__device__ __forceinline__ v8f wmma_bf16(v16bf a, v16bf b, v8f c) {
  // D = A(16x32 bf16) * B(32x16 bf16) + C(16x16 f32)
  return __builtin_amdgcn_wmma_f32_16x16x32_bf16(
      /*neg_a=*/false, a, /*neg_b=*/false, b,
      /*c_mod=*/(short)0, c, /*reuse_a=*/false, /*reuse_b=*/false);
}

// Build a 16x32 A-fragment in contiguous k order (k = 16*half + s): this lane's 16 bf16 values are two
// contiguous 8-element (16B) groups, adjacent in K (16 contiguous K).
__device__ __forceinline__ v16bf ld_a16(const bf16* p) {
  v8bf lo = *(const v8bf*)(p);
  v8bf hi = *(const v8bf*)(p + 8);
  v16bf r;
#pragma unroll
  for (int i = 0; i < 8; ++i) { r[i] = lo[i]; r[i + 8] = hi[i]; }
  return r;
}

// ---------------------------------------------------------------------------
// TDM: DMA a [rows=64, cols=32] bf16 tile (row stride 2048 elems) into LDS.
// D# per CDNA5 ISA; tracked with TENSORcnt.
// ---------------------------------------------------------------------------
__device__ __forceinline__ void tdm_load_tile_64x32(unsigned lds_addr,
                                                    const bf16* gptr) {
  unsigned long long ga = (unsigned long long)(size_t)gptr;
  u32x4 g0;
  g0[0] = 1u;                                    // count=1, user descriptor
  g0[1] = lds_addr;                              // LDS byte address
  g0[2] = (unsigned)(ga & 0xFFFFFFFFu);          // global_addr[31:0]
  g0[3] = (unsigned)((ga >> 32) & 0x01FFFFFFu)   // global_addr[56:32]
          | (2u << 30);                          // type = 2 ("image")
  i32x8 g1;
  g1[0] = 0x00010000;            // workgroup_mask=0, data_size=1 (2 bytes)
  g1[1] = (int)(2048u << 16);    // tensor_dim0 = 2048 (elems)
  g1[2] = (int)(2048u << 16);    // tensor_dim0 hi=0 | tensor_dim1 = 2048
  g1[3] = (int)(32u << 16);      // tensor_dim1 hi=0 | tile_dim0 = 32
  g1[4] = 64;                    // tile_dim1 = 64, tile_dim2 = 0
  g1[5] = 2048;                  // tensor_dim0_stride lo32 (elems)
  g1[6] = 0;                     // stride hi | tensor_dim1_stride lo
  g1[7] = 0;
  i32x4 gz = {};
#if __clang_major__ >= 23
  i32x8 gz8 = {};
  __builtin_amdgcn_tensor_load_to_lds(g0, g1, gz, gz, gz8, 0);
#else
  __builtin_amdgcn_tensor_load_to_lds(g0, g1, gz, gz, 0);
#endif
}

// ---------------------------------------------------------------------------
// fp32 -> bf16 conversion (vectorized by 4)
// ---------------------------------------------------------------------------
__global__ __launch_bounds__(256) void cvt_f32_bf16(const float* __restrict__ in,
                                                    bf16* __restrict__ out, int n) {
  int i = (blockIdx.x * blockDim.x + threadIdx.x) * 4;
  if (i < n) {
    float4 f = *(const float4*)(in + i);
    v4bf o;
    o[0] = (bf16)f.x; o[1] = (bf16)f.y; o[2] = (bf16)f.z; o[3] = (bf16)f.w;
    *(v4bf*)(out + i) = o;
  }
}

// ---------------------------------------------------------------------------
// C = X(2048xD bf16) @ W^T(DxN bf16, W row-major [out,in]) + bias(f32)
// Block = 8 waves = 128(M) x 64(N) tile. B panel (64 rows of W x 32 k) is
// TDM-DMA'd into LDS (double buffered); each wave computes 16x64 with one
// A-fragment reused across 4 WMMAs per k-step.
// OUTMODE: 0 = bf16 row-major, 1 = bf16 transposed (for V), 2 = f32 row-major
// grid = (N/64, T/128); block = 256
// ---------------------------------------------------------------------------
template <int OUTMODE>
__global__ __launch_bounds__(256) void gemm_xwT(const bf16* __restrict__ X,
                                                const bf16* __restrict__ W,
                                                const float* __restrict__ bias,
                                                void* __restrict__ out) {
  __shared__ bf16 bbuf[2][64][32];   // 2 x 4 KB B panels

  const int wv   = threadIdx.x >> 5;
  const int l    = threadIdx.x & 31;
  const int ln   = l & 15;
  const int half = l >> 4;

  const int nbase = blockIdx.x * 64;
  const int mbase = blockIdx.y * 128 + wv * 16;

  const bf16* arow = X + (size_t)(mbase + ln) * DD + (half << 4);
  const bf16* wpan = W + (size_t)nbase * DD;          // 64 rows of W

  const unsigned ldsb = (unsigned)(size_t)(void*)&bbuf[0][0][0];

  v8f acc[4] = {};

  constexpr int NK = DD / 32;
  if (wv == 0) tdm_load_tile_64x32(ldsb, wpan);       // prologue: panel 0

  for (int kb = 0; kb < NK; ++kb) {
    const int cur = kb & 1;
    if (wv == 0) __builtin_amdgcn_s_wait_tensorcnt(0);   // panel `cur` landed
    __syncthreads();
    if (wv == 0 && kb + 1 < NK)                          // overlap next DMA
      tdm_load_tile_64x32(ldsb + (unsigned)(cur ^ 1) * 4096u,
                          wpan + (kb + 1) * 32);

    __builtin_prefetch(arow + (kb + 2) * 32, 0, 0);      // global_prefetch_b8
    v16bf a = ld_a16(arow + kb * 32);
#pragma unroll
    for (int nt = 0; nt < 4; ++nt) {
      v16bf b = *(const v16bf*)&bbuf[cur][nt * 16 + ln][half << 4];
      acc[nt] = wmma_bf16(a, b, acc[nt]);
    }
    __syncthreads();   // all reads of panel `cur` done before it is re-filled
  }

#pragma unroll
  for (int nt = 0; nt < 4; ++nt) {
    const int n = nbase + nt * 16 + ln;
    const float bn = bias[n];
#pragma unroll
    for (int r = 0; r < 8; ++r) {
      const int m = mbase + r + 8 * half;
      const float v = acc[nt][r] + bn;
      if (OUTMODE == 2) {
        ((float*)out)[(size_t)m * DD + n] = v;
      } else if (OUTMODE == 0) {
        ((bf16*)out)[(size_t)m * DD + n] = (bf16)v;
      } else {  // transposed: out is [DD x TT]
        ((bf16*)out)[(size_t)n * TT + m] = (bf16)v;
      }
    }
  }
}

// ---------------------------------------------------------------------------
// Causal flash attention. One wave = one (head, 16-row query tile).
// Q,K: [T x D] bf16 row-major. Vt: [D x T] bf16 (transposed). Y: [T x D] bf16.
// Key blocks of 32; scores = 8 wmma, PV = 8 wmma per block.
// grid = 256 blocks * 8 waves = 2048 waves = H * T/16.
// ---------------------------------------------------------------------------
__global__ __launch_bounds__(256) void attn_causal(const bf16* __restrict__ Q,
                                                   const bf16* __restrict__ K,
                                                   const bf16* __restrict__ Vt,
                                                   bf16* __restrict__ Y) {
  __shared__ float pl[8][16][32];   // per-wave P staging (C-layout -> A-layout)

  const int wv   = threadIdx.x >> 5;
  const int l    = threadIdx.x & 31;
  const int ln   = l & 15;
  const int half = l >> 4;

  const int wid   = blockIdx.x * 8 + wv;
  const int h     = wid >> 7;        // 128 q-tiles per head
  const int qt    = wid & 127;
  const int qbase = qt * 16;

  const float scale = 0.08838834764831845f;  // 1/sqrt(128)

  // Q fragments for this tile: 16 x 128, as 4 A-frags of 16x32
  v16bf qa[4];
  {
    const bf16* qrow = Q + (size_t)(qbase + ln) * DD + h * DH + (half << 4);
#pragma unroll
    for (int kc = 0; kc < 4; ++kc) qa[kc] = ld_a16(qrow + kc * 32);
  }

  v8f o[8] = {};                       // O accumulator: 16 x 128 (8 n-chunks)
  float mr[8], lr[8];
#pragma unroll
  for (int r = 0; r < 8; ++r) { mr[r] = -__builtin_inff(); lr[r] = 0.0f; }

  const int nkb = (qbase + 47) >> 5;   // ceil((qbase+16)/32)
  for (int kb = 0; kb < nkb; ++kb) {
    const int kbase = kb * 32;

    // ---- S = Q @ K^T : two 16x16 score tiles (keys kbase..+15, +16..+31)
    v8f s0 = {}, s1 = {};
#pragma unroll
    for (int kc = 0; kc < 4; ++kc) {
      const bf16* kr0 = K + (size_t)(kbase + ln) * DD + h * DH + kc * 32 + (half << 4);
      const bf16* kr1 = kr0 + (size_t)16 * DD;
      s0 = wmma_bf16(qa[kc], *(const v16bf*)kr0, s0);
      s1 = wmma_bf16(qa[kc], *(const v16bf*)kr1, s1);
    }

    // ---- online softmax (fp32), N is the lane axis within each 16-lane half
    const int key0 = kbase + ln;
    const int key1 = key0 + 16;
#pragma unroll
    for (int r = 0; r < 8; ++r) {
      const int row = qbase + r + 8 * half;
      float v0 = s0[r] * scale;
      float v1 = s1[r] * scale;
      if (key0 > row) v0 = -__builtin_inff();
      if (key1 > row) v1 = -__builtin_inff();

      float mx = fmaxf(v0, v1);
      mx = fmaxf(mx, __shfl_xor(mx, 1, 32));
      mx = fmaxf(mx, __shfl_xor(mx, 2, 32));
      mx = fmaxf(mx, __shfl_xor(mx, 4, 32));
      mx = fmaxf(mx, __shfl_xor(mx, 8, 32));

      const float mnew = fmaxf(mr[r], mx);
      const float p0 = __expf(v0 - mnew);
      const float p1 = __expf(v1 - mnew);

      float sum = p0 + p1;
      sum += __shfl_xor(sum, 1, 32);
      sum += __shfl_xor(sum, 2, 32);
      sum += __shfl_xor(sum, 4, 32);
      sum += __shfl_xor(sum, 8, 32);

      const float alpha = __expf(mr[r] - mnew);
      lr[r] = lr[r] * alpha + sum;
      mr[r] = mnew;
#pragma unroll
      for (int nc = 0; nc < 8; ++nc) o[nc][r] *= alpha;

      pl[wv][r + 8 * half][ln]      = p0;
      pl[wv][r + 8 * half][16 + ln] = p1;
    }

    asm volatile("s_wait_dscnt 0x0" ::: "memory");

    // ---- reload P in A-layout (16x32), convert to bf16
    v16bf pa;
    {
      const int kb8 = half << 4;
#pragma unroll
      for (int i = 0; i < 8; ++i) {
        pa[i]     = (bf16)pl[wv][ln][kb8 + i];
        pa[i + 8] = (bf16)pl[wv][ln][kb8 + 8 + i];
      }
    }

    // ---- O += P @ V : V columns are contiguous in Vt (D x T)
#pragma unroll
    for (int nc = 0; nc < 8; ++nc) {
      const bf16* vr = Vt + (size_t)(h * DH + nc * 16 + ln) * TT + kbase + (half << 4);
      o[nc] = wmma_bf16(pa, *(const v16bf*)vr, o[nc]);
    }
  }

  // ---- epilogue: Y = O / l
#pragma unroll
  for (int r = 0; r < 8; ++r) {
    const float inv = 1.0f / lr[r];
    const int row = qbase + r + 8 * half;
#pragma unroll
    for (int nc = 0; nc < 8; ++nc)
      Y[(size_t)row * DD + h * DH + nc * 16 + ln] = (bf16)(o[nc][r] * inv);
  }
}

// ---------------------------------------------------------------------------
// Host launcher
// ---------------------------------------------------------------------------
extern "C" void kernel_launch(void* const* d_in, const int* in_sizes, int n_in,
                              void* d_out, int out_size, void* d_ws, size_t ws_size,
                              hipStream_t stream) {
  const float* x  = (const float*)d_in[0];
  const float* Wq = (const float*)d_in[1];
  const float* bq = (const float*)d_in[2];
  const float* Wk = (const float*)d_in[3];
  const float* bk = (const float*)d_in[4];
  const float* Wv = (const float*)d_in[5];
  const float* bv = (const float*)d_in[6];
  const float* Wo = (const float*)d_in[7];
  const float* bo = (const float*)d_in[8];

  const size_t NE = (size_t)TT * DD;   // 4M elements per matrix
  bf16* xb  = (bf16*)d_ws;
  bf16* Wqb = xb  + NE;
  bf16* Wkb = Wqb + NE;
  bf16* Wvb = Wkb + NE;
  bf16* Wob = Wvb + NE;
  bf16* Qb  = Wob + NE;
  bf16* Kb  = Qb  + NE;
  bf16* Vtb = Kb  + NE;   // stored transposed [D x T]
  bf16* Yb  = Vtb + NE;

  // fp32 -> bf16 converts (each matrix: 4M elems, 4 per thread)
  {
    dim3 g((unsigned)(NE / 4 + 255) / 256), b(256);
    cvt_f32_bf16<<<g, b, 0, stream>>>(x,  xb,  (int)NE);
    cvt_f32_bf16<<<g, b, 0, stream>>>(Wq, Wqb, (int)NE);
    cvt_f32_bf16<<<g, b, 0, stream>>>(Wk, Wkb, (int)NE);
    cvt_f32_bf16<<<g, b, 0, stream>>>(Wv, Wvb, (int)NE);
    cvt_f32_bf16<<<g, b, 0, stream>>>(Wo, Wob, (int)NE);
  }

  // QKV projections (bf16 WMMA, fp32 accum + bias)
  {
    dim3 g(DD / 64, TT / 128), b(256);
    gemm_xwT<0><<<g, b, 0, stream>>>(xb, Wqb, bq, (void*)Qb);
    gemm_xwT<0><<<g, b, 0, stream>>>(xb, Wkb, bk, (void*)Kb);
    gemm_xwT<1><<<g, b, 0, stream>>>(xb, Wvb, bv, (void*)Vtb);  // V transposed
  }

  // causal flash attention: 2048 waves, 8 per block
  attn_causal<<<dim3(256), dim3(256), 0, stream>>>(Qb, Kb, Vtb, Yb);

  // output projection, fp32 result
  gemm_xwT<2><<<dim3(DD / 64, TT / 128), dim3(256), 0, stream>>>(Yb, Wob, bo, d_out);
}
